// BitInstantaneousFrequencyHead_35605278884132
// MI455X (gfx1250) — 8-bit matrix operand order repaired and checked by reading, compiled, not run on hardware
//
#include <hip/hip_runtime.h>
#include <stdint.h>

#define B_DIM   8
#define T_DIM   4096
#define D_IN    1024
#define D_OUT   1024
#define NTOK    (B_DIM * T_DIM)
#define PI_F    3.14159265358979323846f
#define TWO_PI  6.28318530717958647692f

typedef __attribute__((ext_vector_type(8))) int v8i;

#define WQ_OFF    ((size_t)0)
#define XQ_OFF    (WQ_OFF + (size_t)D_OUT * D_IN)
#define SX_OFF    (XQ_OFF + (size_t)NTOK * D_IN)
#define PART_OFF  (SX_OFF + (size_t)NTOK * sizeof(float))
#define SCAL_OFF  (PART_OFF + (size_t)D_OUT * sizeof(float))

__device__ __forceinline__ void async_b128(uint32_t lds_off, const void* gaddr) {
    asm volatile("global_load_async_to_lds_b128 %0, %1, off"
                 :: "v"(lds_off), "v"(gaddr) : "memory");
}
#if defined(__has_builtin) && __has_builtin(__builtin_amdgcn_s_wait_asynccnt)
#define WAIT_ASYNC(n) __builtin_amdgcn_s_wait_asynccnt(n)
#else
#define WAIT_ASYNC(n) asm volatile("s_wait_asynccnt %0" :: "n"(n) : "memory")
#endif

__global__ void k_row_abssum(const float* __restrict__ W, float* __restrict__ partial) {
    __shared__ float s[256];
    const int row = blockIdx.x;
    const float* p = W + (size_t)row * D_IN;
    float acc = 0.f;
    for (int i = threadIdx.x; i < D_IN; i += 256) acc += fabsf(p[i]);
    s[threadIdx.x] = acc;
    __syncthreads();
    for (int off = 128; off > 0; off >>= 1) {
        if (threadIdx.x < off) s[threadIdx.x] += s[threadIdx.x + off];
        __syncthreads();
    }
    if (threadIdx.x == 0) partial[row] = s[0];
}

__global__ void k_finalize_wscale(const float* __restrict__ partial, float* __restrict__ scal) {
    __shared__ float s[256];
    float acc = 0.f;
    for (int i = threadIdx.x; i < D_OUT; i += 256) acc += partial[i];
    s[threadIdx.x] = acc;
    __syncthreads();
    for (int off = 128; off > 0; off >>= 1) {
        if (threadIdx.x < off) s[threadIdx.x] += s[threadIdx.x + off];
        __syncthreads();
    }
    if (threadIdx.x == 0) {
        float mean = s[0] / (float)((size_t)D_OUT * D_IN);
        scal[0] = fmaxf(mean, 1e-5f);
    }
}

__global__ void k_quant_w(const float* __restrict__ W, const float* __restrict__ scal,
                          int8_t* __restrict__ wq) {
    const int i4 = blockIdx.x * 256 + threadIdx.x;
    const float wscale = 1.f / scal[0];
    const float4 w = ((const float4*)W)[i4];
    int q0 = (int)fminf(fmaxf(rintf(w.x * wscale), -1.f), 1.f);
    int q1 = (int)fminf(fmaxf(rintf(w.y * wscale), -1.f), 1.f);
    int q2 = (int)fminf(fmaxf(rintf(w.z * wscale), -1.f), 1.f);
    int q3 = (int)fminf(fmaxf(rintf(w.w * wscale), -1.f), 1.f);
    uint32_t packed = (uint32_t)(q0 & 255) | ((uint32_t)(q1 & 255) << 8) |
                      ((uint32_t)(q2 & 255) << 16) | ((uint32_t)(q3 & 255) << 24);
    ((uint32_t*)wq)[i4] = packed;
}

__global__ void k_quant_x(const float* __restrict__ x, int8_t* __restrict__ xq,
                          float* __restrict__ inv_sx) {
    __shared__ float s[256];
    const int tok = blockIdx.x;
    const float4 v = ((const float4*)(x + (size_t)tok * D_IN))[threadIdx.x];
    float m = fmaxf(fmaxf(fabsf(v.x), fabsf(v.y)), fmaxf(fabsf(v.z), fabsf(v.w)));
    s[threadIdx.x] = m;
    __syncthreads();
    for (int off = 128; off > 0; off >>= 1) {
        if (threadIdx.x < off) s[threadIdx.x] = fmaxf(s[threadIdx.x], s[threadIdx.x + off]);
        __syncthreads();
    }
    const float amax  = fmaxf(s[0], 1e-5f);
    const float scale = 127.f / amax;
    if (threadIdx.x == 0) inv_sx[tok] = amax / 127.f;
    int q0 = (int)fminf(fmaxf(rintf(v.x * scale), -128.f), 127.f);
    int q1 = (int)fminf(fmaxf(rintf(v.y * scale), -128.f), 127.f);
    int q2 = (int)fminf(fmaxf(rintf(v.z * scale), -128.f), 127.f);
    int q3 = (int)fminf(fmaxf(rintf(v.w * scale), -128.f), 127.f);
    uint32_t packed = (uint32_t)(q0 & 255) | ((uint32_t)(q1 & 255) << 8) |
                      ((uint32_t)(q2 & 255) << 16) | ((uint32_t)(q3 & 255) << 24);
    ((uint32_t*)(xq + (size_t)tok * D_IN))[threadIdx.x] = packed;
}

#define ROWB 80
__global__ void __launch_bounds__(256)
k_gemm_iu8(const int8_t* __restrict__ xq, const int8_t* __restrict__ wq,
           const float* __restrict__ inv_sx, const float* __restrict__ scal,
           const float* __restrict__ bias, const float* __restrict__ cwp,
           float* __restrict__ out) {
    __shared__ int8_t sA[2][128 * ROWB];
    __shared__ int8_t sB[2][128 * ROWB];

    const int tid  = threadIdx.x;
    const int wid  = tid >> 5;
    const int lane = tid & 31;
    const int half = lane >> 4;
    const int r    = lane & 15;
    const int m_blk = blockIdx.x & ((NTOK / 128) - 1);
    const int n_blk = blockIdx.x >> 8;
    const int m0 = m_blk * 128;
    const int n0 = n_blk * 128;

    const int c0   = tid * 2;
    const int row0 = c0 >> 2, off0 = (c0 & 3) * 16;
    const int row1 = row0,    off1 = off0 + 16;
    const int8_t* gA = xq + (size_t)m0 * D_IN;
    const int8_t* gB = wq + (size_t)n0 * D_IN;

    auto issue = [&](int buf, int k0) {
        async_b128((uint32_t)(uintptr_t)&sA[buf][row0 * ROWB + off0],
                   gA + (size_t)row0 * D_IN + k0 + off0);
        async_b128((uint32_t)(uintptr_t)&sA[buf][row1 * ROWB + off1],
                   gA + (size_t)row1 * D_IN + k0 + off1);
        async_b128((uint32_t)(uintptr_t)&sB[buf][row0 * ROWB + off0],
                   gB + (size_t)row0 * D_IN + k0 + off0);
        async_b128((uint32_t)(uintptr_t)&sB[buf][row1 * ROWB + off1],
                   gB + (size_t)row1 * D_IN + k0 + off1);
    };

    v8i acc[8] = {};

    issue(0, 0);
    issue(1, 64);

    for (int kk = 0; kk < 16; ++kk) {
        const int buf = kk & 1;
        if (kk < 15) { WAIT_ASYNC(4); } else { WAIT_ASYNC(0); }
        __syncthreads();

        const int8_t* abase = &sA[buf][(wid * 16 + r) * ROWB + (half << 4)];
        v8i a;
#pragma unroll
        for (int j = 0; j < 4; ++j) {
            const int2 d = *(const int2*)(abase + 8 * (j & 1) + 32 * (j >> 1));
            a[2 * j]     = d.x;
            a[2 * j + 1] = d.y;
        }
        v8i bt[8];
#pragma unroll
        for (int s = 0; s < 8; ++s) {
            const int8_t* bc = &sB[buf][(s * 16 + r) * ROWB + (half << 4)];
            const int4 b0 = *(const int4*)(bc);
            const int4 b1 = *(const int4*)(bc + 32);
            bt[s][0] = b0.x; bt[s][1] = b0.y; bt[s][2] = b0.z; bt[s][3] = b0.w;
            bt[s][4] = b1.x; bt[s][5] = b1.y; bt[s][6] = b1.z; bt[s][7] = b1.w;
        }
#pragma unroll
        for (int s = 0; s < 8; ++s) {
            acc[s] = __builtin_amdgcn_wmma_i32_16x16x64_iu8(
                true, a, true, bt[s], acc[s], false, false);
        }
        __syncthreads();
        if (kk < 14) issue(buf, (kk + 2) * 64);
    }

    const float inv_sw = scal[0];
    const float cw     = cwp[0];
    float bn[8];
#pragma unroll
    for (int s = 0; s < 8; ++s) bn[s] = bias[n0 + s * 16 + r];

    const int mw0 = m0 + wid * 16;
#pragma unroll
    for (int j = 0; j < 8; ++j) {
        const int m = mw0 + j + (half << 3);
        const float fm = inv_sx[m] * inv_sw;
#pragma unroll
        for (int s = 0; s < 8; ++s) {
            const int n = n0 + s * 16 + r;
            const float val = (float)acc[s][j] * fm + bn[s];
            out[(size_t)m * D_OUT + n] = tanhf(val) * PI_F * cw;
        }
    }
}

__global__ void k_scan(float* __restrict__ out) {
    const int idx = blockIdx.x * blockDim.x + threadIdx.x;
    const int b = idx / D_OUT;
    const int d = idx % D_OUT;
    float* p = out + (size_t)b * T_DIM * D_OUT + d;
    float acc = 0.f;
#pragma unroll 8
    for (int t = 0; t < T_DIM; ++t) {
        acc += p[(size_t)t * D_OUT];
        float r = fmodf(acc + PI_F, TWO_PI);
        if (r < 0.f) r += TWO_PI;
        p[(size_t)t * D_OUT] = r - PI_F;
    }
}

extern "C" void kernel_launch(void* const* d_in, const int* in_sizes, int n_in,
                              void* d_out, int out_size, void* d_ws, size_t ws_size,
                              hipStream_t stream) {
    const float* x  = (const float*)d_in[0];
    const float* W  = (const float*)d_in[1];
    const float* b  = (const float*)d_in[2];
    const float* cw = (const float*)d_in[3];
    float* out = (float*)d_out;

    char* ws = (char*)d_ws;
    int8_t* wq     = (int8_t*)(ws + WQ_OFF);
    int8_t* xq     = (int8_t*)(ws + XQ_OFF);
    float*  inv_sx = (float*)(ws + SX_OFF);
    float*  part   = (float*)(ws + PART_OFF);
    float*  scal   = (float*)(ws + SCAL_OFF);

    (void)in_sizes; (void)n_in; (void)out_size; (void)ws_size;

    k_row_abssum<<<D_OUT, 256, 0, stream>>>(W, part);
    k_finalize_wscale<<<1, 256, 0, stream>>>(part, scal);
    k_quant_w<<<(D_OUT * D_IN / 4) / 256, 256, 0, stream>>>(W, scal, wq);
    k_quant_x<<<NTOK, 256, 0, stream>>>(x, xq, inv_sx);
    k_gemm_iu8<<<(NTOK / 128) * (D_OUT / 128), 256, 0, stream>>>(
        xq, wq, inv_sx, scal, b, cw, out);
    k_scan<<<(B_DIM * D_OUT) / 256, 256, 0, stream>>>(out);
}
